// RNNModel_52106543235715
// MI455X (gfx1250) — matrix operand order repaired and checked by reading, not compiled, not run on hardware
//
#include <hip/hip_runtime.h>
#include <math.h>

// ---------------------------------------------------------------------------
// RNNModel (B=64, T=512, I=256, H=1024, L=4) for MI455X / gfx1250 (wave32).
//
// Strategy:
//  * Input projection per layer: bf16 WMMA GEMM (v_wmma_f32_16x16x32_bf16),
//    P = x @ W_ih^T + b_ih + b_hh, f32 output.
//  * Recurrence: persistent kernel, 64 blocks, W_hh slice resident in LDS
//    (~132KB of the 320KB/WGP), per-step h tile staged into LDS with
//    GLOBAL_LOAD_ASYNC_TO_LDS_B128 (ASYNCcnt), one grid-wide atomic barrier
//    per timestep, h state ping-pong in global (128KB, L2 resident).
//  * Final FC: wave-per-row bf16 dot with wave32 shfl reduction.
// ---------------------------------------------------------------------------

typedef __attribute__((ext_vector_type(16))) __bf16 v16bf;
typedef __attribute__((ext_vector_type(8)))  float  v8f;

union Frag16 {
    v16bf v;
    uint4 q[2];
    unsigned short u[16];
};

__device__ __forceinline__ unsigned short f32_to_bf16_rne(float f) {
    unsigned u = __float_as_uint(f);
    unsigned r = u + 0x7FFFu + ((u >> 16) & 1u);
    return (unsigned short)(r >> 16);
}
__device__ __forceinline__ float bf16_to_f32(unsigned short s) {
    return __uint_as_float(((unsigned)s) << 16);
}

// Async 16B copy global -> LDS (CDNA5, tracked by ASYNCcnt).
// Low 32 bits of a generic pointer into LDS are the LDS byte address
// (ISA 10.2 aperture rules), which is exactly what VDST wants.
__device__ __forceinline__ void async_ld_b128(void* ldsDst, const void* gSrc) {
    unsigned ldsAddr = (unsigned)(uintptr_t)ldsDst;
    asm volatile("global_load_async_to_lds_b128 %0, %1, off"
                 :: "v"(ldsAddr), "v"(gSrc) : "memory");
}
__device__ __forceinline__ void wait_async0() {
    asm volatile("s_wait_asynccnt 0x0" ::: "memory");
}

// ---------------------------------------------------------------- cast kernel
__global__ void cast_f32_bf16(const float* __restrict__ src,
                              unsigned short* __restrict__ dst, int n) {
    int i = blockIdx.x * blockDim.x + threadIdx.x;
    int stride = gridDim.x * blockDim.x;
    for (; i < n; i += stride) dst[i] = f32_to_bf16_rne(src[i]);
}

// ------------------------------------------------------------- GEMM (xW+bias)
// A: [M,K] bf16 row-major (activations), W: [N,K] bf16 row-major (torch W_ih),
// P[m,n] = sum_k A[m,k]*W[n,k] + b1[n] + b2[n].  One wave per 16x16 tile.
__global__ void __launch_bounds__(256)
gemm_xw_bias(const unsigned short* __restrict__ A,
             const unsigned short* __restrict__ W,
             const float* __restrict__ b1, const float* __restrict__ b2,
             float* __restrict__ P, int M, int N, int K) {
    const int lane = threadIdx.x & 31;
    const int wave = threadIdx.x >> 5;
    const int ntN  = N >> 4;
    const int gid  = blockIdx.x * 8 + wave;
    const int tM   = gid / ntN;
    const int tN   = gid % ntN;
    if (tM * 16 >= M) return;

    const int hiK  = lane >> 4;                  // lane group 0/1
    const int l16  = lane & 15;
    // A fragment: lanes 0-15 hold K=[0..7]+[16..23], lanes 16-31 K=[8..15]+[24..31]
    const unsigned short* aPtr = A + (size_t)(tM * 16 + l16) * K + hiK * 8;
    // B fragment: lanes 0-15 hold K=[0..7]+[16..23], lanes 16-31 K=[8..15]+[24..31]
    const unsigned short* wPtr = W + (size_t)(tN * 16 + l16) * K + hiK * 8;

    v8f c0 = {}; v8f c1 = {};
    const int ks = K >> 5;                       // 8 (layer0) or 32
    for (int kb = 0; kb < ks; kb += 2) {
        Frag16 a, b;
        a.q[0] = *(const uint4*)(aPtr + kb * 32);
        a.q[1] = *(const uint4*)(aPtr + kb * 32 + 16);
        b.q[0] = *(const uint4*)(wPtr + kb * 32);
        b.q[1] = *(const uint4*)(wPtr + kb * 32 + 16);
        c0 = __builtin_amdgcn_wmma_f32_16x16x32_bf16(false, a.v, false, b.v,
                                                     (short)0, c0, false, false);
        Frag16 a2, b2;
        a2.q[0] = *(const uint4*)(aPtr + (kb + 1) * 32);
        a2.q[1] = *(const uint4*)(aPtr + (kb + 1) * 32 + 16);
        b2.q[0] = *(const uint4*)(wPtr + (kb + 1) * 32);
        b2.q[1] = *(const uint4*)(wPtr + (kb + 1) * 32 + 16);
        c1 = __builtin_amdgcn_wmma_f32_16x16x32_bf16(false, a2.v, false, b2.v,
                                                     (short)0, c1, false, false);
    }
    v8f c = c0 + c1;
    const int n    = tN * 16 + l16;
    const float bb = b1[n] + b2[n];
#pragma unroll
    for (int r = 0; r < 8; ++r) {
        int m = tM * 16 + r + hiK * 8;           // C/D layout: vgpr r -> M=r(+8)
        P[(size_t)m * N + n] = c[r] + bb;
    }
}

// ----------------------------------------------------------- grid-wide barrier
__device__ __forceinline__ void gridBarrier(volatile unsigned* cnt,
                                            volatile unsigned* gen,
                                            unsigned nblk) {
    __syncthreads();
    if (threadIdx.x == 0) {
        __threadfence();                              // release my stores (device)
        unsigned g = *gen;
        unsigned arrived = atomicAdd((unsigned*)cnt, 1u);
        if (arrived == nblk - 1) {
            *cnt = 0;
            __threadfence();
            atomicAdd((unsigned*)gen, 1u);
        } else {
            while (*gen == g) { __builtin_amdgcn_s_sleep(1); }
        }
        __threadfence();                              // acquire others' stores
    }
    __syncthreads();
}

// ------------------------------------------------------------ recurrence step
// Persistent kernel. 64 blocks x 128 threads. Block (mt,ng) owns batch rows
// [mt*16, +16) and cols [ng*64, +64). W_hh slice (64 x 1024 bf16) stays in LDS
// for all 512 steps; per step the 16x1024 bf16 h tile is staged into LDS with
// async-to-LDS b128 copies.
__global__ void __launch_bounds__(128)
rnn_recur(const unsigned short* __restrict__ Whh,   // [H,H] bf16, row n, col k
          const float* __restrict__ P,              // [B*T, H] f32
          unsigned short* __restrict__ hs,          // [B*T, H] bf16 out
          unsigned short* __restrict__ h0buf,       // [B,H] bf16 ping
          unsigned short* __restrict__ h1buf,       // [B,H] bf16 pong
          unsigned* __restrict__ bar) {             // [0]=count, [1]=gen
    constexpr int H  = 1024;
    constexpr int T  = 512;
    constexpr int NB = 64;                          // grid blocks
    constexpr int LK = H + 8;                       // padded halfs per LDS row

    extern __shared__ unsigned short lds[];
    unsigned short* Alds = lds;                     // [16][LK]   (h tile)
    unsigned short* Blds = lds + 16 * LK;           // [64][LK]   (W slice)

    const int tid  = threadIdx.x;
    const int lane = tid & 31;
    const int wave = tid >> 5;
    const int mt   = blockIdx.x >> 4;               // 0..3   batch-row tile
    const int ng   = blockIdx.x & 15;               // 0..15  64-col group
    const int m0   = mt * 16;
    const int n0   = ng * 64;

    // Stage W_hh^T slice once: B(k, n) = W_hh[n, k]; rows n-local, contig k.
    for (int idx = tid; idx < 64 * (H / 8); idx += 128) {
        int n  = idx >> 7;                          // H/8 == 128
        int kc = (idx & 127) << 3;
        async_ld_b128(&Blds[n * LK + kc], &Whh[(size_t)(n0 + n) * H + kc]);
    }
    // h_0 = 0 (each block zeros exactly its own output region).
    for (int idx = tid; idx < 16 * 64; idx += 128) {
        int r = idx >> 6, cc = idx & 63;
        h0buf[(size_t)(m0 + r) * H + n0 + cc] = 0;
    }
    gridBarrier(bar, bar + 1, NB);

    unsigned short* hcur = h0buf;
    unsigned short* hnxt = h1buf;

    const int hiK  = lane >> 4;
    const int l16  = lane & 15;
    const int nloc = wave * 16 + l16;
    const int ncol = n0 + nloc;
    const unsigned short* aBase = Alds + l16 * LK + hiK * 8;
    const unsigned short* bBase = Blds + nloc * LK + hiK * 8;

    for (int t = 0; t < T; ++t) {
        // Stage h_{t-1}[m0..m0+15, 0..H) into LDS (32KB) via async DMA.
        for (int idx = tid; idx < 16 * (H / 8); idx += 128) {
            int m  = idx >> 7;
            int kc = (idx & 127) << 3;
            async_ld_b128(&Alds[m * LK + kc], &hcur[(size_t)(m0 + m) * H + kc]);
        }
        wait_async0();            // covers W-slice copies on first iteration too
        __syncthreads();

        v8f c0 = {}, c1 = {}, c2 = {}, c3 = {};
#pragma unroll
        for (int kb = 0; kb < H / 32; kb += 4) {
            Frag16 a, b;
            a.q[0] = *(const uint4*)(aBase + kb * 32);
            a.q[1] = *(const uint4*)(aBase + kb * 32 + 16);
            b.q[0] = *(const uint4*)(bBase + kb * 32);
            b.q[1] = *(const uint4*)(bBase + kb * 32 + 16);
            c0 = __builtin_amdgcn_wmma_f32_16x16x32_bf16(false, a.v, false, b.v,
                                                         (short)0, c0, false, false);
            Frag16 a1, b1;
            a1.q[0] = *(const uint4*)(aBase + (kb + 1) * 32);
            a1.q[1] = *(const uint4*)(aBase + (kb + 1) * 32 + 16);
            b1.q[0] = *(const uint4*)(bBase + (kb + 1) * 32);
            b1.q[1] = *(const uint4*)(bBase + (kb + 1) * 32 + 16);
            c1 = __builtin_amdgcn_wmma_f32_16x16x32_bf16(false, a1.v, false, b1.v,
                                                         (short)0, c1, false, false);
            Frag16 a2, b2;
            a2.q[0] = *(const uint4*)(aBase + (kb + 2) * 32);
            a2.q[1] = *(const uint4*)(aBase + (kb + 2) * 32 + 16);
            b2.q[0] = *(const uint4*)(bBase + (kb + 2) * 32);
            b2.q[1] = *(const uint4*)(bBase + (kb + 2) * 32 + 16);
            c2 = __builtin_amdgcn_wmma_f32_16x16x32_bf16(false, a2.v, false, b2.v,
                                                         (short)0, c2, false, false);
            Frag16 a3, b3;
            a3.q[0] = *(const uint4*)(aBase + (kb + 3) * 32);
            a3.q[1] = *(const uint4*)(aBase + (kb + 3) * 32 + 16);
            b3.q[0] = *(const uint4*)(bBase + (kb + 3) * 32);
            b3.q[1] = *(const uint4*)(bBase + (kb + 3) * 32 + 16);
            c3 = __builtin_amdgcn_wmma_f32_16x16x32_bf16(false, a3.v, false, b3.v,
                                                         (short)0, c3, false, false);
        }
        v8f c = (c0 + c1) + (c2 + c3);

#pragma unroll
        for (int r = 0; r < 8; ++r) {
            int    brow = m0 + r + hiK * 8;
            size_t pidx = ((size_t)brow * T + t) * H + ncol;
            float  v    = c[r] + P[pidx];
            unsigned short hb = f32_to_bf16_rne(tanhf(v));
            hnxt[(size_t)brow * H + ncol] = hb;     // state for step t+1
            hs[pidx] = hb;                          // layer output
        }
        // One barrier/step: my reads of hcur happened before arrive; writes to
        // that same buffer (as hnxt of step t+1) happen only after release.
        gridBarrier(bar, bar + 1, NB);
        unsigned short* tmp = hcur; hcur = hnxt; hnxt = tmp;
    }
}

// ------------------------------------------------------------------- final FC
__global__ void __launch_bounds__(256)
fc_out(const unsigned short* __restrict__ hsbf, const float* __restrict__ w,
       const float* __restrict__ bptr, float* __restrict__ out, int rows, int H) {
    int gw   = (int)((blockIdx.x * blockDim.x + threadIdx.x) >> 5);
    int lane = threadIdx.x & 31;
    if (gw >= rows) return;
    const unsigned short* row = hsbf + (size_t)gw * H;
    float s = 0.f;
    for (int h = lane; h < H; h += 32) s += bf16_to_f32(row[h]) * w[h];
#pragma unroll
    for (int off = 16; off > 0; off >>= 1) s += __shfl_xor(s, off, 32);
    if (lane == 0) out[gw] = s + bptr[0];
}

// ------------------------------------------------------------------- launcher
extern "C" void kernel_launch(void* const* d_in, const int* in_sizes, int n_in,
                              void* d_out, int out_size, void* d_ws, size_t ws_size,
                              hipStream_t stream) {
    (void)in_sizes; (void)n_in; (void)out_size; (void)ws_size;
    const int B = 64, T = 512, I = 256, H = 1024, L = 4;
    const size_t M = (size_t)B * T;

    char* ws = (char*)d_ws;
    size_t off = 0;
    auto alloc = [&](size_t bytes) {
        void* p = ws + off;
        off = (off + bytes + 255) & ~(size_t)255;
        return p;
    };
    unsigned short* actA = (unsigned short*)alloc(M * H * 2);        // 64 MB
    unsigned short* actB = (unsigned short*)alloc(M * H * 2);        // 64 MB
    float*          P    = (float*)alloc(M * H * 4);                 // 128 MB
    unsigned short* Wih  = (unsigned short*)alloc((size_t)H * H * 2);// 2 MB
    unsigned short* Whh  = (unsigned short*)alloc((size_t)H * H * 2);// 2 MB
    unsigned short* h0   = (unsigned short*)alloc((size_t)B * H * 2);
    unsigned short* h1   = (unsigned short*)alloc((size_t)B * H * 2);
    unsigned*       bar  = (unsigned*)alloc(256);

    hipMemsetAsync(bar, 0, 256, stream);   // deterministic barrier state

    // x -> bf16 into actA
    cast_f32_bf16<<<512, 256, 0, stream>>>((const float*)d_in[0], actA, B * T * I);

    const size_t shmem = (size_t)(16 + 64) * (H + 8) * 2;  // ~161 KB dyn LDS

    for (int l = 0; l < L; ++l) {
        const int Din = (l == 0) ? I : H;
        const float* W_ih = (const float*)d_in[1 + 4 * l + 0];
        const float* W_hh = (const float*)d_in[1 + 4 * l + 1];
        const float* b_ih = (const float*)d_in[1 + 4 * l + 2];
        const float* b_hh = (const float*)d_in[1 + 4 * l + 3];

        cast_f32_bf16<<<512, 256, 0, stream>>>(W_ih, Wih, H * Din);
        cast_f32_bf16<<<512, 256, 0, stream>>>(W_hh, Whh, H * H);

        unsigned short* Ain  = (l & 1) ? actB : actA;
        unsigned short* Aout = (l & 1) ? actA : actB;

        const int tiles  = (int)(M / 16) * (H / 16);       // 131072
        const int blocks = tiles / 8;
        gemm_xw_bias<<<blocks, 256, 0, stream>>>(Ain, Wih, b_ih, b_hh, P,
                                                 (int)M, H, Din);
        rnn_recur<<<64, 128, shmem, stream>>>(Whh, P, Aout, h0, h1, bar);
    }

    const float* fcw = (const float*)d_in[1 + 4 * L];
    const float* fcb = (const float*)d_in[2 + 4 * L];
    unsigned short* hsFinal = ((L - 1) & 1) ? actA : actB; // layer 3 wrote actA
    fc_out<<<(int)(M / 8), 256, 0, stream>>>(hsFinal, fcw, fcb,
                                             (float*)d_out, (int)M, H);
}
